// SequenceLSTM_79998060855656
// MI455X (gfx1250) — compile-verified, matrix operand order checked by reading, not run on hardware
//
#include <hip/hip_runtime.h>
#include <hip/hip_bf16.h>
#include <stdint.h>

// ---------------------------------------------------------------------------
// 2-layer LSTM (B=4096, T=256, H=256, I=5) + FC(H->1) for MI455X / gfx1250.
//
//  * Persistent batch tiles: grid = B/16 = 256 workgroups x 128 threads
//    (4 wave32). Each WG owns 16 batch rows for ALL timesteps -> no global
//    sync across the sequential T loop.
//  * All GEMMs via v_wmma_f32_16x16x32_bf16. Weights pre-packed into the
//    WMMA B-fragment layout; streamed from L2 (192MB) as coalesced b128
//    loads each step. Pointer-laundering via empty asm blocks LICM from
//    hoisting 256 timesteps worth of fragments into registers (which
//    previously caused mass scratch spills).
//  * h state kept in LDS (bf16, padded stride) as the shared A operand;
//    cell state c kept in VGPRs; gate math f32 VALU co-executes with WMMA.
//  * Layer 1 stages h1 input tiles via global_load_async_to_lds_b128
//    (ASYNCcnt) double-buffered one timestep ahead; s_wait_asynccnt.
// ---------------------------------------------------------------------------

#define LSTM_I 5
#define LSTM_H 256
#define LSTM_B 4096
#define LSTM_T 256
#define G4H    1024     // 4*H gate width
#define NT4H   64       // N-tiles (16 wide) across 4H
#define HSTR   264      // padded LDS row stride in bf16 elems (banks spread)
#define ROWB   (HSTR*2) // 528 bytes per padded LDS row

typedef __attribute__((ext_vector_type(16))) __bf16 v16bf;
typedef __attribute__((ext_vector_type(8)))  float  v8f;

union FragU { v16bf v; int4 q[2]; };

__device__ __forceinline__ float sigmoidf_(float x) {
  return 1.0f / (1.0f + __expf(-x));
}

// Async global->LDS copy, 16B per lane (ASYNCcnt path, CDNA5).
__device__ __forceinline__ void async_cp16(uint32_t lds_byte_off, const void* gaddr) {
  asm volatile("global_load_async_to_lds_b128 %0, %1, off"
               :: "v"(lds_byte_off), "v"(gaddr) : "memory");
}
__device__ __forceinline__ void wait_async0() {
  asm volatile("s_wait_asynccnt 0" ::: "memory");
}

// Value-launder a uniform pointer through an SGPR pair so loads based on it
// cannot be hoisted out of loops (LICM/CSE barrier, zero instructions).
__device__ __forceinline__ const __bf16* launder(const __bf16* p) {
  asm volatile("" : "+s"(p));
  return p;
}

// A fragment (16x32 bf16) from LDS: lane l -> row l&15, K = kc*32 + (l>=16)*16..
__device__ __forceinline__ v16bf load_frag_lds(const __bf16* base, int strideElems,
                                               int kc, int lane) {
  const int row = lane & 15;
  const int kb  = kc * 32 + ((lane >> 4) << 4);
  const __bf16* p = base + row * strideElems + kb;
  FragU u;
  u.q[0] = *(const int4*)(p);
  u.q[1] = *(const int4*)(p + 8);
  return u.v;
}

// B fragment (32x16 bf16) from pre-packed global weights: 1KB per (kc, ntile)
// block, 32B contiguous per lane -> two fully-coalesced b128 loads (L2 hits).
__device__ __forceinline__ v16bf load_frag_glb(const __bf16* pack, int kc, int nt,
                                               int lane) {
  const __bf16* p = pack + (((size_t)(kc * NT4H + nt) * 32 + lane) << 4);
  FragU u;
  u.q[0] = *(const int4*)(p);
  u.q[1] = *(const int4*)(p + 8);
  return u.v;
}

// ---------------------------------------------------------------------------
// Prep: pack f32 weights [4H, Ksrc] into bf16 WMMA B-fragment blocks.
// Block (kc, nt): lane l holds column n = nt*16 + (l&15),
// K = kc*32 + (l>=16)*16 + j, j = 0..15. K beyond Ksrc zero-padded.
// grid = KC*64 blocks x 32 threads.
// ---------------------------------------------------------------------------
__global__ void pack_wmma_b(const float* __restrict__ src, __bf16* __restrict__ dst,
                            int Ksrc) {
  const int bx   = blockIdx.x;
  const int kc   = bx >> 6;
  const int nt   = bx & 63;
  const int lane = threadIdx.x;
  const int n  = nt * 16 + (lane & 15);
  const int kb = kc * 32 + ((lane >> 4) << 4);
  __bf16* d = dst + (((size_t)bx * 32 + lane) << 4);
#pragma unroll
  for (int j = 0; j < 16; ++j) {
    const int k = kb + j;
    const float v = (k < Ksrc) ? src[(size_t)n * Ksrc + k] : 0.0f;
    d[j] = (__bf16)v;
  }
}

__global__ void bias_sum_kernel(const float* __restrict__ a,
                                const float* __restrict__ b,
                                float* __restrict__ o) {
  const int i = blockIdx.x * blockDim.x + threadIdx.x;
  if (i < G4H) o[i] = a[i] + b[i];
}

// ---------------------------------------------------------------------------
// Fused LSTM layer. LAYER==0: input is x f32 [B,T,5] (padded K=32 GEMM),
// writes h1 bf16 [B,T,H]. LAYER==1: input is h1 bf16 (async-staged, K=256
// GEMM), emits FC output f32 [B,T].
// 128 threads, wave w owns h-columns [w*64, w*64+64) across all 4 gates.
// ---------------------------------------------------------------------------
template<int LAYER>
__global__ __launch_bounds__(128)
void lstm_layer_kernel(const float*  __restrict__ xin,
                       const __bf16* __restrict__ hin,
                       const __bf16* __restrict__ wihP,
                       const __bf16* __restrict__ whhP,
                       const float*  __restrict__ biasSum,
                       __bf16*       __restrict__ hout,
                       const float*  __restrict__ fcw,
                       const float*  __restrict__ fcb,
                       float*        __restrict__ out) {
  constexpr int KCIN     = (LAYER == 0) ? 1 : (LSTM_H / 32);
  constexpr int INSTRIDE = (LAYER == 0) ? 32 : HSTR;
  constexpr int INELEMS  = (LAYER == 0) ? (16 * 32) : (2 * 16 * HSTR);

  __shared__ __align__(16) __bf16 hbuf[16 * HSTR];   // h_t tile (A operand)
  __shared__ __align__(16) __bf16 inbuf[INELEMS];    // input tile (dbl-buf L1)
  __shared__ float sbias[G4H];
  __shared__ float sfcw[LSTM_H];
  __shared__ float sfcpart[16 * 8];

  const int tid  = threadIdx.x;
  const int wave = tid >> 5;
  const int lane = tid & 31;
  const int b0   = blockIdx.x * 16;

  // ---- init ----
  for (int i = tid; i < 16 * HSTR; i += 128) hbuf[i] = (__bf16)0.0f;
  if (LAYER == 0) {
    for (int i = tid; i < 16 * 32; i += 128) inbuf[i] = (__bf16)0.0f;
  }
  for (int i = tid; i < G4H; i += 128) sbias[i] = biasSum[i];
  if (LAYER == 1) {
    for (int i = tid; i < LSTM_H; i += 128) sfcw[i] = fcw[i];
  }
  const float fcb0 = (LAYER == 1) ? fcb[0] : 0.0f;

  v8f cst[4];                       // cell state: 16 rows x 64 cols per wave
#pragma unroll
  for (int s = 0; s < 4; ++s)
#pragma unroll
    for (int v = 0; v < 8; ++v) cst[s][v] = 0.0f;

  if (LAYER == 1) {
    // prefetch t=0 input tile into buffer 0 (async -> LDS)
    const uint32_t ldsbase = (uint32_t)(uintptr_t)(&inbuf[0]);
#pragma unroll
    for (int k2 = 0; k2 < 4; ++k2) {
      const int c = tid + 128 * k2;
      const int r = c >> 5, c16 = c & 31;
      const char* g = (const char*)hin +
          (((size_t)(b0 + r) * LSTM_T + 0) * LSTM_H) * 2 + c16 * 16;
      async_cp16(ldsbase + (uint32_t)(r * ROWB + c16 * 16), g);
    }
  }
  __syncthreads();

  const int rowbase = ((lane >> 4) << 3);

  for (int t = 0; t < LSTM_T; ++t) {
    // ---- stage input tile ----
    if (LAYER == 0) {
      if (tid < 16) {
        const float* xp = xin + ((size_t)(b0 + tid) * LSTM_T + t) * LSTM_I;
#pragma unroll
        for (int i = 0; i < LSTM_I; ++i) inbuf[tid * 32 + i] = (__bf16)xp[i];
      }
      __syncthreads();
    } else {
      wait_async0();               // tile for step t is now in LDS
      __syncthreads();
      if (t + 1 < LSTM_T) {        // prefetch t+1 into the other buffer
        const int bufN = (t + 1) & 1;
        const uint32_t ldsbase = (uint32_t)(uintptr_t)(&inbuf[bufN * 16 * HSTR]);
#pragma unroll
        for (int k2 = 0; k2 < 4; ++k2) {
          const int c = tid + 128 * k2;
          const int r = c >> 5, c16 = c & 31;
          const char* g = (const char*)hin +
              (((size_t)(b0 + r) * LSTM_T + (t + 1)) * LSTM_H) * 2 + c16 * 16;
          async_cp16(ldsbase + (uint32_t)(r * ROWB + c16 * 16), g);
        }
      }
    }

    // ---- gate pre-activations: acc = [x_t | h1_t] @ W_ih^T + h_{t-1} @ W_hh^T
    v8f acc[16];
#pragma unroll
    for (int i = 0; i < 16; ++i)
#pragma unroll
      for (int v = 0; v < 8; ++v) acc[i][v] = 0.0f;

    const __bf16* inbase =
        (LAYER == 0) ? inbuf : (inbuf + (t & 1) * 16 * HSTR);

#pragma unroll
    for (int kc = 0; kc < KCIN; ++kc) {
      const v16bf a = load_frag_lds(inbase, INSTRIDE, kc, lane);
#pragma unroll
      for (int g = 0; g < 4; ++g) {
        const __bf16* wp = launder(wihP);   // block LICM: keep B in L2, not VGPRs
#pragma unroll
        for (int s = 0; s < 4; ++s) {
          const v16bf b = load_frag_glb(wp, kc, g * 16 + wave * 4 + s, lane);
          acc[g * 4 + s] = __builtin_amdgcn_wmma_f32_16x16x32_bf16(
              false, a, false, b, (short)0, acc[g * 4 + s], false, false);
        }
      }
    }
#pragma unroll
    for (int kc = 0; kc < LSTM_H / 32; ++kc) {
      const v16bf a = load_frag_lds(hbuf, HSTR, kc, lane);
#pragma unroll
      for (int g = 0; g < 4; ++g) {
        const __bf16* wp = launder(whhP);
#pragma unroll
        for (int s = 0; s < 4; ++s) {
          const v16bf b = load_frag_glb(wp, kc, g * 16 + wave * 4 + s, lane);
          acc[g * 4 + s] = __builtin_amdgcn_wmma_f32_16x16x32_bf16(
              false, a, false, b, (short)0, acc[g * 4 + s], false, false);
        }
      }
    }

    __syncthreads();   // all waves done reading hbuf for this step

    // ---- elementwise cell/gate update (torch gate order i,f,g,o) ----
#pragma unroll
    for (int s = 0; s < 4; ++s) {
      const int col = wave * 64 + s * 16 + (lane & 15);
      const float bI = sbias[col];
      const float bF = sbias[256 + col];
      const float bG = sbias[512 + col];
      const float bO = sbias[768 + col];
#pragma unroll
      for (int v = 0; v < 8; ++v) {
        const float ig = sigmoidf_(acc[0 + s][v] + bI);
        const float fg = sigmoidf_(acc[4 + s][v] + bF);
        const float gg = tanhf(acc[8 + s][v] + bG);
        const float og = sigmoidf_(acc[12 + s][v] + bO);
        const float cn = fg * cst[s][v] + ig * gg;
        cst[s][v] = cn;
        const float hv = og * tanhf(cn);
        hbuf[(size_t)(rowbase + v) * HSTR + col] = (__bf16)hv;
      }
    }
    __syncthreads();   // h_t visible to all waves

    // ---- emit ----
    if (LAYER == 0) {
      // h1[b, t, :] <- hbuf (coalesced b128 copies)
#pragma unroll
      for (int k2 = 0; k2 < 4; ++k2) {
        const int c = tid + 128 * k2;
        const int r = c >> 5, c16 = c & 31;
        const int4 vv = *(const int4*)((const char*)hbuf + r * ROWB + c16 * 16);
        *(int4*)((char*)hout +
                 (((size_t)(b0 + r) * LSTM_T + t) * LSTM_H) * 2 + c16 * 16) = vv;
      }
    } else {
      // FC: out[b, t] = h2[b,t,:] . fc_w + fc_b (two-phase reduction)
      const int r = tid >> 3, seg = tid & 7;
      float sum = 0.0f;
#pragma unroll 8
      for (int j = 0; j < 32; ++j) {
        const int col = seg * 32 + j;
        sum += (float)hbuf[(size_t)r * HSTR + col] * sfcw[col];
      }
      sfcpart[r * 8 + seg] = sum;
      __syncthreads();
      if (tid < 16) {
        float o = fcb0;
#pragma unroll
        for (int j = 0; j < 8; ++j) o += sfcpart[tid * 8 + j];
        out[(size_t)(b0 + tid) * LSTM_T + t] = o;
      }
    }
  }
}

// ---------------------------------------------------------------------------
extern "C" void kernel_launch(void* const* d_in, const int* in_sizes, int n_in,
                              void* d_out, int out_size, void* d_ws, size_t ws_size,
                              hipStream_t stream) {
  const float* x     = (const float*)d_in[0];
  const float* w_ih0 = (const float*)d_in[1];
  const float* w_hh0 = (const float*)d_in[2];
  const float* b_ih0 = (const float*)d_in[3];
  const float* b_hh0 = (const float*)d_in[4];
  const float* w_ih1 = (const float*)d_in[5];
  const float* w_hh1 = (const float*)d_in[6];
  const float* b_ih1 = (const float*)d_in[7];
  const float* b_hh1 = (const float*)d_in[8];
  const float* fc_w  = (const float*)d_in[9];
  const float* fc_b  = (const float*)d_in[10];
  float* out = (float*)d_out;

  // workspace layout
  char* ws = (char*)d_ws;
  size_t off = 0;
  __bf16* h1 = (__bf16*)(ws + off);
  off += (size_t)LSTM_B * LSTM_T * LSTM_H * 2;              // 512 MB
  __bf16* wih0p = (__bf16*)(ws + off); off += (size_t)1 * NT4H * 32 * 16 * 2;
  __bf16* whh0p = (__bf16*)(ws + off); off += (size_t)8 * NT4H * 32 * 16 * 2;
  __bf16* wih1p = (__bf16*)(ws + off); off += (size_t)8 * NT4H * 32 * 16 * 2;
  __bf16* whh1p = (__bf16*)(ws + off); off += (size_t)8 * NT4H * 32 * 16 * 2;
  float* bias0 = (float*)(ws + off); off += G4H * sizeof(float);
  float* bias1 = (float*)(ws + off); off += G4H * sizeof(float);

  // prep: pack weights into WMMA B-fragment layout; fold bias pairs
  pack_wmma_b<<<1 * NT4H, 32, 0, stream>>>(w_ih0, wih0p, LSTM_I);
  pack_wmma_b<<<8 * NT4H, 32, 0, stream>>>(w_hh0, whh0p, LSTM_H);
  pack_wmma_b<<<8 * NT4H, 32, 0, stream>>>(w_ih1, wih1p, LSTM_H);
  pack_wmma_b<<<8 * NT4H, 32, 0, stream>>>(w_hh1, whh1p, LSTM_H);
  bias_sum_kernel<<<(G4H + 255) / 256, 256, 0, stream>>>(b_ih0, b_hh0, bias0);
  bias_sum_kernel<<<(G4H + 255) / 256, 256, 0, stream>>>(b_ih1, b_hh1, bias1);

  // layer 0: x -> h1 (persistent batch tiles across all T)
  lstm_layer_kernel<0><<<LSTM_B / 16, 128, 0, stream>>>(
      x, nullptr, wih0p, whh0p, bias0, h1, nullptr, nullptr, nullptr);

  // layer 1 + FC: h1 -> out
  lstm_layer_kernel<1><<<LSTM_B / 16, 128, 0, stream>>>(
      nullptr, h1, wih1p, whh1p, bias1, nullptr, fc_w, fc_b, out);

  (void)in_sizes; (void)n_in; (void)out_size; (void)ws_size;
}
